// ImageCNN_69827578298495
// MI455X (gfx1250) — matrix operand order repaired and checked by reading, not compiled, not run on hardware
//
#include <hip/hip_runtime.h>

// ---------------------------------------------------------------------------
// ImageCNN on MI455X (gfx1250, wave32):
//   conv5x5(3->16) + bias + relu + maxpool2x2  -> bf16-packed features (B,416)
//   linear (B,400)@(400,1000)+b via v_wmma_f32_16x16x32_bf16 (f32 accumulate)
// ---------------------------------------------------------------------------

typedef __attribute__((ext_vector_type(16))) __bf16        v16bf;
typedef __attribute__((ext_vector_type(8)))  float         v8f;
typedef __attribute__((ext_vector_type(4)))  unsigned int  u32x4;

#define KREAL  400
#define KPAD   416        // 13 * 32
#define KSTEPS 13
#define NOUT   1000
#define NPAD   1024       // 64 tiles of 16

__device__ __forceinline__ unsigned short f2bf(float f) {
    // round-to-nearest-even fp32 -> bf16
    unsigned int u = __float_as_uint(f);
    unsigned int r = u + 0x7FFFu + ((u >> 16) & 1u);
    return (unsigned short)(r >> 16);
}

// ---------------------------------------------------------------------------
// Kernel 1: pack lin_w (1000 x 400, row-major fp32) into bf16 B-fragments.
// Fragment (nt, ks) is the 32(K) x 16(N) tile:  n = nt*16 + 0..15,
// k = ks*32 + 0..31 (zero-padded past K=400 / N=1000).
// Per-lane layout (CDNA5 ISA 7.12.2, 16-bit B): lane L<16 -> col L, K {0..7,16..23};
// lane L>=16 -> col L-16, K {8..15,24..31}; 16 bf16 per lane stored contiguously.
// Flat index: ((nt*13 + ks)*32 + lane)*16 + e
// ---------------------------------------------------------------------------
__global__ __launch_bounds__(256) void pack_w_kernel(
    const float* __restrict__ lw, unsigned short* __restrict__ Wf)
{
    int id = blockIdx.x * 256 + threadIdx.x;      // 64*13*32 = 26624 threads
    int lane = id & 31;
    int ks   = (id >> 5) % KSTEPS;
    int nt   = id / (32 * KSTEPS);
    int n    = nt * 16 + (lane & 15);
    int kb   = ks * 32 + ((lane >> 4) << 3);      // {0..7,16..23} or {8..15,24..31}
    unsigned short* o = Wf + (long)id * 16;
#pragma unroll
    for (int e = 0; e < 16; ++e) {
        int k = kb + e + ((e >> 3) << 3);
        float v = (n < NOUT && k < KREAL) ? lw[(long)n * KREAL + k] : 0.0f;
        o[e] = f2bf(v);
    }
}

// ---------------------------------------------------------------------------
// Kernel 2: conv + bias + relu + 2x2 maxpool, emit bf16 rows of length 416.
// Block = 256 threads = 8 waves; one image per wave, staged in LDS.
// Lane (0..24) owns pooled position (ph,pw); 4x16 fp32 accumulators.
// relu(max2x2(acc)+bias) == max2x2(relu(acc+bias)) since relu is monotone.
// ---------------------------------------------------------------------------
__global__ __launch_bounds__(256) void conv_pool_pack_kernel(
    const float* __restrict__ fmap, const float* __restrict__ cw,
    const float* __restrict__ cb, unsigned short* __restrict__ Xp)
{
    __shared__ float wsh[16 * 3 * 25];            // 1200 floats
    __shared__ float img[8 * 3 * 196];            // 8 images * 588 floats

    int tid = threadIdx.x;
    for (int i = tid; i < 1200; i += 256) wsh[i] = cw[i];
    long gbase = (long)blockIdx.x * (8 * 588);
    for (int i = tid; i < 8 * 588; i += 256) img[i] = fmap[gbase + i];
    __syncthreads();

    int wave = tid >> 5, lane = tid & 31;
    long b = (long)blockIdx.x * 8 + wave;
    unsigned short* xrow = Xp + b * KPAD;

    if (lane < 16) xrow[KREAL + lane] = 0;        // zero K padding 400..415

    if (lane < 25) {
        int ph = lane / 5, pw = lane % 5;
        float a0[16], a1[16], a2[16], a3[16];
#pragma unroll
        for (int co = 0; co < 16; ++co) { a0[co]=0.f; a1[co]=0.f; a2[co]=0.f; a3[co]=0.f; }

        const float* ib = &img[wave * 588];
        for (int ci = 0; ci < 3; ++ci) {
            const float* ip = ib + ci * 196;
            const float* wp = &wsh[ci * 25];
            for (int ky = 0; ky < 5; ++ky) {
                int y = 2 * ph + ky;
                for (int kx = 0; kx < 5; ++kx) {
                    int x = 2 * pw + kx;
                    float i00 = ip[y * 14 + x];
                    float i01 = ip[y * 14 + x + 1];
                    float i10 = ip[y * 14 + 14 + x];
                    float i11 = ip[y * 14 + 14 + x + 1];
#pragma unroll
                    for (int co = 0; co < 16; ++co) {
                        float wv = wp[co * 75 + ky * 5 + kx];
                        a0[co] = fmaf(wv, i00, a0[co]);
                        a1[co] = fmaf(wv, i01, a1[co]);
                        a2[co] = fmaf(wv, i10, a2[co]);
                        a3[co] = fmaf(wv, i11, a3[co]);
                    }
                }
            }
        }
#pragma unroll
        for (int co = 0; co < 16; ++co) {
            float m = fmaxf(fmaxf(a0[co], a1[co]), fmaxf(a2[co], a3[co]));
            float v = fmaxf(m + cb[co], 0.0f);
            xrow[co * 25 + lane] = f2bf(v);       // flatten order = c*25+ph*5+pw
        }
    }
}

// ---------------------------------------------------------------------------
// Kernel 3: GEMM  out(B x 1000) = X(B x 416)bf16 @ Wf + lin_b  via WMMA.
// Block = 8 waves arranged 4M x 2N -> 128x128 block tile.
// Wave tile = 32M x 64N: 2 A-frags + 4 B-frags -> 8 WMMAs per k-step
// (1.5 b128 loads per WMMA). 13 k-steps over K=416.
// A layout (ISA 7.12.2, 16-bit A): lane L<16 -> row L, K {0..7,16..23};
// lane L>=16 -> row L-16, K {8..15,24..31}  => two b128 loads per fragment.
// ---------------------------------------------------------------------------
__global__ __launch_bounds__(256) void gemm_bf16_kernel(
    const unsigned short* __restrict__ Xp, const unsigned short* __restrict__ Wf,
    const float* __restrict__ lb, float* __restrict__ out)
{
    int tid  = threadIdx.x;
    int wave = tid >> 5, lane = tid & 31;
    int nb = blockIdx.x & 7;                      // 8 n-blocks of 128 (Npad=1024)
    int mb = blockIdx.x >> 3;                     // 256 m-blocks of 128
    int m0 = mb * 128 + (wave >> 1) * 32;
    int n0 = nb * 128 + (wave & 1) * 64;

    int half = (lane >> 4) & 1;                   // lower/upper half-wave
    int aoff = half * 8;                          // K half-chunk offset for A
    int arow = lane & 15;

    const unsigned short* xbase0 = Xp + (long)(m0 + arow) * KPAD + aoff;
    const unsigned short* xbase1 = xbase0 + 16L * KPAD;
    int nt0 = n0 >> 4;
    // four consecutive 16-col B fragment streams
    const unsigned short* wb[4];
#pragma unroll
    for (int nt = 0; nt < 4; ++nt)
        wb[nt] = Wf + (((long)(nt0 + nt) * KSTEPS) * 32 + lane) * 16;

    v8f acc[2][4];
#pragma unroll
    for (int mt = 0; mt < 2; ++mt)
#pragma unroll
        for (int nt = 0; nt < 4; ++nt) acc[mt][nt] = v8f{};

    for (int ks = 0; ks < KSTEPS; ++ks) {
        int k0 = ks * 32;
        union Frag { v16bf v; u32x4 u[2]; };
        Frag a[2], bfr[4];
        a[0].u[0] = *(const u32x4*)(xbase0 + k0);
        a[0].u[1] = *(const u32x4*)(xbase0 + k0 + 16);
        a[1].u[0] = *(const u32x4*)(xbase1 + k0);
        a[1].u[1] = *(const u32x4*)(xbase1 + k0 + 16);
#pragma unroll
        for (int nt = 0; nt < 4; ++nt) {
            const u32x4* pb = (const u32x4*)(wb[nt] + (long)ks * (32 * 16));
            bfr[nt].u[0] = pb[0];
            bfr[nt].u[1] = pb[1];
        }

        // pull next A chunk into WGP$ (locality 3 -> WGP-scope prefetch)
        __builtin_prefetch(xbase0 + k0 + 32, 0, 3);
        __builtin_prefetch(xbase1 + k0 + 32, 0, 3);

#pragma unroll
        for (int mt = 0; mt < 2; ++mt)
#pragma unroll
            for (int nt = 0; nt < 4; ++nt)
                acc[mt][nt] = __builtin_amdgcn_wmma_f32_16x16x32_bf16(
                    false, a[mt].v, false, bfr[nt].v, (short)0, acc[mt][nt],
                    false, false);
    }

    // Epilogue: C/D layout -> lane = N column, VGPR r = M row (+8 upper half)
    int colbase = n0 + (lane & 15);
    int rowbase = m0 + half * 8;
#pragma unroll
    for (int nt = 0; nt < 4; ++nt) {
        int col = colbase + nt * 16;
        if (col < NOUT) {
            float bias = lb[col];
#pragma unroll
            for (int mt = 0; mt < 2; ++mt) {
                float* o = out + (long)(rowbase + mt * 16) * NOUT + col;
#pragma unroll
                for (int r = 0; r < 8; ++r)
                    o[(long)r * NOUT] = acc[mt][nt][r] + bias;
            }
        }
    }
}

// ---------------------------------------------------------------------------
extern "C" void kernel_launch(void* const* d_in, const int* in_sizes, int n_in,
                              void* d_out, int out_size, void* d_ws, size_t ws_size,
                              hipStream_t stream) {
    const float* fmap = (const float*)d_in[0];   // (B,3,14,14)
    const float* cw   = (const float*)d_in[1];   // (16,3,5,5)
    const float* cb   = (const float*)d_in[2];   // (16,)
    const float* lw   = (const float*)d_in[3];   // (1000,400)
    const float* lb   = (const float*)d_in[4];   // (1000,)
    float*       out  = (float*)d_out;           // (B,1000)

    int B = in_sizes[0] / (3 * 14 * 14);         // 32768

    // workspace: X packed bf16 [B][416], then W fragments [64][13][32][16] bf16
    unsigned short* Xp = (unsigned short*)d_ws;
    unsigned short* Wf = Xp + (size_t)B * KPAD;

    pack_w_kernel<<<(64 * KSTEPS * 32) / 256, 256, 0, stream>>>(lw, Wf);
    conv_pool_pack_kernel<<<B / 8, 256, 0, stream>>>(fmap, cw, cb, Xp);
    gemm_bf16_kernel<<<(B / 128) * (NPAD / 128), 256, 0, stream>>>(Xp, Wf, lb, out);
}
